// HEAT_57775900066118
// MI455X (gfx1250) — compile-verified, matrix operand order checked by reading, not run on hardware
//
#include <hip/hip_runtime.h>
#include <hip/hip_bf16.h>

typedef __attribute__((ext_vector_type(16))) _Float16 v16h;
typedef __attribute__((ext_vector_type(8)))  _Float16 h8;
typedef __attribute__((ext_vector_type(8)))  float    v8f;
typedef __attribute__((ext_vector_type(4)))  float    f4;

union V16 { v16h v; h8 h[2]; };

#define SLOPE 0.2f

__device__ __forceinline__ float lrelu(float v) { return v >= 0.f ? v : SLOPE * v; }

__device__ __forceinline__ void atomicMaxF(float* addr, float val) {
    unsigned int* ua = (unsigned int*)addr;
    unsigned int old = __float_as_uint(-3.402823466e38f);
    old = atomicCAS(ua, old, __float_as_uint(val)); // fast path vs init value
    while (__uint_as_float(old) < val) {
        unsigned int assumed = old;
        old = atomicCAS(ua, assumed, __float_as_uint(val));
        if (old == assumed) break;
    }
}

// sum over the 16-lane half-wave group (WMMA column group)
__device__ __forceinline__ float redux16(float v) {
    v += __shfl_xor(v, 1);
    v += __shfl_xor(v, 2);
    v += __shfl_xor(v, 4);
    v += __shfl_xor(v, 8);
    return v;
}

// ---------------------------------------------------------------------------
// init: zero out[], denom[], amax[] = -inf, att_ete[t] = lrelu(ete_emb[t]).w
// ---------------------------------------------------------------------------
__global__ __launch_bounds__(256) void heat_init(float* __restrict__ out,
                                                 float* __restrict__ amax,
                                                 float* __restrict__ denom,
                                                 float* __restrict__ att_ete,
                                                 const float* __restrict__ ete_emb,
                                                 const float* __restrict__ att_w,
                                                 int N) {
    int i = blockIdx.x * blockDim.x + threadIdx.x;
    if (i < N * 32) out[i] = 0.f;
    if (i < N) { amax[i] = -3.402823466e38f; denom[i] = 0.f; }
    if (i < 8) {
        float s = 0.f;
#pragma unroll
        for (int j = 0; j < 16; ++j) s += lrelu(ete_emb[i * 16 + j]) * att_w[64 + j];
        att_ete[i] = s;
    }
}

// ---------------------------------------------------------------------------
// HeteroLinear + per-node attention partials:
//   xl = x @ W[node_type] + b[node_type];  ai = xl.w[0:32];  aj = xl.w[32:64]
// grid-stride over 16-node tiles, B (4 types x 2 halves) register-resident
// ---------------------------------------------------------------------------
__global__ __launch_bounds__(256) void heat_hetero(const float* __restrict__ x,
                                                   const float* __restrict__ hetero_w,
                                                   const float* __restrict__ hetero_b,
                                                   const float* __restrict__ att_w,
                                                   const int* __restrict__ node_type,
                                                   float* __restrict__ xl,
                                                   _Float16* __restrict__ xlh,
                                                   float* __restrict__ ai,
                                                   float* __restrict__ aj,
                                                   int N, int tiles) {
    const int lane  = threadIdx.x & 31;
    const int wave  = threadIdx.x >> 5;
    const int wpb   = blockDim.x >> 5;
    const int khalf = (lane >> 4) << 4;            // 0 or 16
    const int col16 = lane & 15;
    const int stride = gridDim.x * wpb;

    // register-cache B: hetero_w[t][:, h*16+col16] along K
    v16h bw[4][2];
#pragma unroll
    for (int t = 0; t < 4; ++t)
#pragma unroll
        for (int h = 0; h < 2; ++h)
#pragma unroll
            for (int kk = 0; kk < 16; ++kk)
                bw[t][h][kk] =
                    (_Float16)hetero_w[(t * 32 + khalf + kk) * 32 + h * 16 + col16];

    const float awi[2] = {att_w[col16], att_w[16 + col16]};
    const float awj[2] = {att_w[32 + col16], att_w[48 + col16]};

    for (int tile = blockIdx.x * wpb + wave; tile < tiles; tile += stride) {
        const int base  = tile * 16;
        const int node  = base + col16;
        const int nodec = node < N ? node : N - 1;
        const int rowbase = base + ((lane >> 4) << 3);

        // A tile (16x32 f16): vector loads of this lane's 16 contiguous floats
        V16 A;
        {
            const f4* xr = (const f4*)(x + (size_t)nodec * 32 + khalf);
            f4 q0 = xr[0], q1 = xr[1], q2 = xr[2], q3 = xr[3];
#pragma unroll
            for (int kk = 0; kk < 4; ++kk) {
                A.v[kk]      = (_Float16)q0[kk];
                A.v[4 + kk]  = (_Float16)q1[kk];
                A.v[8 + kk]  = (_Float16)q2[kk];
                A.v[12 + kk] = (_Float16)q3[kk];
            }
        }

        float aacc[8], jacc[8];
#pragma unroll
        for (int r = 0; r < 8; ++r) { aacc[r] = 0.f; jacc[r] = 0.f; }

#pragma unroll
        for (int h = 0; h < 2; ++h) {
            v8f acc = {0.f, 0.f, 0.f, 0.f, 0.f, 0.f, 0.f, 0.f};
#pragma unroll
            for (int t = 0; t < 4; ++t) {
                v8f c = {0.f, 0.f, 0.f, 0.f, 0.f, 0.f, 0.f, 0.f};
                c = __builtin_amdgcn_wmma_f32_16x16x32_f16(false, A.v, false,
                                                           bw[t][h], (short)0, c,
                                                           false, false);
#pragma unroll
                for (int r = 0; r < 8; ++r) {
                    int nd  = rowbase + r;
                    int ndc = nd < N ? nd : N - 1;
                    if (node_type[ndc] == t) acc[r] = c[r];
                }
            }
#pragma unroll
            for (int r = 0; r < 8; ++r) {
                int nd = rowbase + r;
                if (nd < N) {
                    int c2  = h * 16 + col16;
                    float v = acc[r] + hetero_b[node_type[nd] * 32 + c2];
                    xl[nd * 32 + c2]  = v;
                    xlh[nd * 32 + c2] = (_Float16)v;
                    aacc[r] += v * awi[h];
                    jacc[r] += v * awj[h];
                }
            }
        }
        // per-row attention partials: reduce over the 16-lane column group
#pragma unroll
        for (int r = 0; r < 8; ++r) {
            float a = redux16(aacc[r]);
            float j = redux16(jacc[r]);
            int nd = rowbase + r;
            if (col16 == 0 && nd < N) { ai[nd] = a; aj[nd] = j; }
        }
    }
}

// ---------------------------------------------------------------------------
// eae = lrelu(edge_attr @ edge_attr_w); also eadot[e] = eae[e].att_w[80:96]
// ---------------------------------------------------------------------------
__global__ __launch_bounds__(256) void heat_eae(const float* __restrict__ edge_attr,
                                                const float* __restrict__ eaw,
                                                const float* __restrict__ att_w,
                                                _Float16* __restrict__ eaeh,
                                                float* __restrict__ eadot,
                                                int E, int tiles) {
    const int lane  = threadIdx.x & 31;
    const int wave  = threadIdx.x >> 5;
    const int wpb   = blockDim.x >> 5;
    const int khalf = (lane >> 4) << 4;
    const int col16 = lane & 15;
    const int stride = gridDim.x * wpb;

    v16h bw;
#pragma unroll
    for (int kk = 0; kk < 16; ++kk) {
        int k = khalf + kk;
        bw[kk] = (k < 16) ? (_Float16)eaw[k * 16 + col16] : (_Float16)0.f;
    }
    const float aw = att_w[80 + col16];

    for (int tile = blockIdx.x * wpb + wave; tile < tiles; tile += stride) {
        const int base = tile * 16;
        const int erow = base + col16;
        const int erc  = erow < E ? erow : E - 1;

        V16 A;
        A.v = v16h{};
        if (khalf == 0) {                          // K=0..15 holds the 16 attrs
            const f4* ar = (const f4*)(edge_attr + (size_t)erc * 16);
            f4 q0 = ar[0], q1 = ar[1], q2 = ar[2], q3 = ar[3];
#pragma unroll
            for (int kk = 0; kk < 4; ++kk) {
                A.v[kk]      = (_Float16)q0[kk];
                A.v[4 + kk]  = (_Float16)q1[kk];
                A.v[8 + kk]  = (_Float16)q2[kk];
                A.v[12 + kk] = (_Float16)q3[kk];
            }
        }
        v8f c = {0.f, 0.f, 0.f, 0.f, 0.f, 0.f, 0.f, 0.f};
        c = __builtin_amdgcn_wmma_f32_16x16x32_f16(false, A.v, false, bw, (short)0,
                                                   c, false, false);
        const int rowbase = base + ((lane >> 4) << 3);
#pragma unroll
        for (int r = 0; r < 8; ++r) {
            int e   = rowbase + r;
            float v = lrelu(c[r]);
            if (e < E) eaeh[e * 16 + col16] = (_Float16)v;
            float d = redux16(v * aw);
            if (col16 == 0 && e < E) eadot[e] = d;
        }
    }
}

// ---------------------------------------------------------------------------
// Attention logits + segment max (thread per edge, factored dot products)
// ---------------------------------------------------------------------------
__global__ __launch_bounds__(256) void heat_alpha(const float* __restrict__ ai,
                                                  const float* __restrict__ aj,
                                                  const float* __restrict__ eadot,
                                                  const float* __restrict__ att_ete,
                                                  const int* __restrict__ edge_index,
                                                  const int* __restrict__ edge_type,
                                                  float* __restrict__ wbuf,
                                                  float* __restrict__ amax, int E) {
    int e = blockIdx.x * blockDim.x + threadIdx.x;
    if (e >= E) return;
    __builtin_prefetch(&edge_index[e + 4096], 0, 0);   // global_prefetch_b8
    int s  = edge_index[e];
    int d  = edge_index[E + e];
    int et = edge_type[e];
    float acc = lrelu(ai[d] + aj[s] + att_ete[et] + eadot[e]);
    wbuf[e] = acc;
    atomicMaxF(&amax[d], acc);
}

// ---------------------------------------------------------------------------
// exp(logit - max) + segment sum (in-place on wbuf)
// ---------------------------------------------------------------------------
__global__ __launch_bounds__(256) void heat_expsum(const int* __restrict__ edge_index,
                                                   const float* __restrict__ amax,
                                                   float* __restrict__ wbuf,
                                                   float* __restrict__ denom, int E) {
    int e = blockIdx.x * blockDim.x + threadIdx.x;
    if (e >= E) return;
    int d = edge_index[E + e];
    float v = expf(wbuf[e] - amax[d]);
    wbuf[e] = v;
    atomicAdd(&denom[d], v);
}

// ---------------------------------------------------------------------------
// Message GEMM [x_j | eae | 0] (16x64) @ lin_w (64x32 padded) + weighted scatter
// grid-stride, B (2 chunks x 2 halves) register-resident
// ---------------------------------------------------------------------------
__global__ __launch_bounds__(256) void heat_msg(const _Float16* __restrict__ xlh,
                                                const _Float16* __restrict__ eaeh,
                                                const float* __restrict__ lin_w,
                                                const int* __restrict__ edge_index,
                                                const float* __restrict__ wbuf,
                                                const float* __restrict__ denom,
                                                float* __restrict__ out,
                                                int E, int tiles) {
    const int lane  = threadIdx.x & 31;
    const int wave  = threadIdx.x >> 5;
    const int wpb   = blockDim.x >> 5;
    const int khalf = (lane >> 4) << 4;
    const int col16 = lane & 15;
    const int stride = gridDim.x * wpb;

    // register-cache B: lin_w (48x32, K padded to 64), chunk c covers K=c*32..c*32+31
    v16h bw[2][2];
#pragma unroll
    for (int kk = 0; kk < 16; ++kk) {
        int k0 = khalf + kk;
        int k1 = 32 + khalf + kk;
#pragma unroll
        for (int h = 0; h < 2; ++h) {
            bw[0][h][kk] = (_Float16)lin_w[k0 * 32 + h * 16 + col16];
            bw[1][h][kk] = (k1 < 48) ? (_Float16)lin_w[k1 * 32 + h * 16 + col16]
                                     : (_Float16)0.f;
        }
    }

    for (int tile = blockIdx.x * wpb + wave; tile < tiles; tile += stride) {
        const int base = tile * 16;
        const int erow = base + col16;
        const int erc  = erow < E ? erow : E - 1;
        const int srcn = edge_index[erc];

        V16 A0, A1;
        {   // chunk0: x_j (16 contiguous f16 = 2x16B vector loads)
            const h8* sr = (const h8*)(xlh + (size_t)srcn * 32 + khalf);
            A0.h[0] = sr[0];
            A0.h[1] = sr[1];
        }
        A1.v = v16h{};
        if (khalf == 0) {   // chunk1: K=32..47 -> eae row
            const h8* er = (const h8*)(eaeh + (size_t)erc * 16);
            A1.h[0] = er[0];
            A1.h[1] = er[1];
        }

        const int rowbase = base + ((lane >> 4) << 3);
        float al[8];
        int   dstv[8];
#pragma unroll
        for (int r = 0; r < 8; ++r) {
            int e   = rowbase + r;
            int ec  = e < E ? e : E - 1;
            int d   = edge_index[E + ec];
            dstv[r] = d;
            al[r]   = wbuf[ec] / (denom[d] + 1e-16f);
        }

#pragma unroll
        for (int h = 0; h < 2; ++h) {
            v8f c = {0.f, 0.f, 0.f, 0.f, 0.f, 0.f, 0.f, 0.f};
            c = __builtin_amdgcn_wmma_f32_16x16x32_f16(false, A0.v, false, bw[0][h],
                                                       (short)0, c, false, false);
            c = __builtin_amdgcn_wmma_f32_16x16x32_f16(false, A1.v, false, bw[1][h],
                                                       (short)0, c, false, false);
#pragma unroll
            for (int r = 0; r < 8; ++r) {
                if (rowbase + r < E)
                    atomicAdd(&out[dstv[r] * 32 + h * 16 + col16], c[r] * al[r]);
            }
        }
    }
}

// ---------------------------------------------------------------------------
extern "C" void kernel_launch(void* const* d_in, const int* in_sizes, int n_in,
                              void* d_out, int out_size, void* d_ws, size_t ws_size,
                              hipStream_t stream) {
    const float* x          = (const float*)d_in[0];
    const float* edge_attr  = (const float*)d_in[1];
    const float* hetero_w   = (const float*)d_in[2];
    const float* hetero_b   = (const float*)d_in[3];
    const float* ete_emb    = (const float*)d_in[4];
    const float* eaw        = (const float*)d_in[5];
    const float* att_w      = (const float*)d_in[6];
    const float* lin_w      = (const float*)d_in[7];
    const int*   edge_index = (const int*)d_in[8];
    const int*   node_type  = (const int*)d_in[9];
    const int*   edge_type  = (const int*)d_in[10];

    const int N = in_sizes[0] / 32;
    const int E = in_sizes[8] / 2;
    float* out = (float*)d_out;

    // workspace partition (256B aligned)
    char* p = (char*)d_ws;
    auto up = [](size_t v) { return (v + 255) & ~(size_t)255; };
    float*    xl      = (float*)p;    p += up((size_t)N * 32 * 4);
    _Float16* xlh     = (_Float16*)p; p += up((size_t)N * 32 * 2);
    _Float16* eaeh    = (_Float16*)p; p += up((size_t)E * 16 * 2);
    float*    wbuf    = (float*)p;    p += up((size_t)E * 4);
    float*    amax    = (float*)p;    p += up((size_t)N * 4);
    float*    denom   = (float*)p;    p += up((size_t)N * 4);
    float*    ai      = (float*)p;    p += up((size_t)N * 4);
    float*    aj      = (float*)p;    p += up((size_t)N * 4);
    float*    eadot   = (float*)p;    p += up((size_t)E * 4);
    float*    att_ete = (float*)p;    p += up((size_t)8 * 4);

    const int T = 256;
    heat_init<<<(N * 32 + T - 1) / T, T, 0, stream>>>(out, amax, denom, att_ete,
                                                      ete_emb, att_w, N);

    const int nodeTiles = (N + 15) / 16;
    heat_hetero<<<(nodeTiles + 15) / 16, T, 0, stream>>>(x, hetero_w, hetero_b,
                                                         att_w, node_type, xl, xlh,
                                                         ai, aj, N, nodeTiles);

    const int edgeTiles = (E + 15) / 16;
    heat_eae<<<(edgeTiles + 31) / 32, T, 0, stream>>>(edge_attr, eaw, att_w, eaeh,
                                                      eadot, E, edgeTiles);

    heat_alpha<<<(E + T - 1) / T, T, 0, stream>>>(ai, aj, eadot, att_ete,
                                                  edge_index, edge_type, wbuf,
                                                  amax, E);

    heat_expsum<<<(E + T - 1) / T, T, 0, stream>>>(edge_index, amax, wbuf, denom, E);

    heat_msg<<<(edgeTiles + 31) / 32, T, 0, stream>>>(xlh, eaeh, lin_w, edge_index,
                                                      wbuf, denom, out, E, edgeTiles);
}
